// DeepAR_57372173140071
// MI455X (gfx1250) — 8-bit matrix operand order repaired and checked by reading, compiled, not run on hardware
//
#include <hip/hip_runtime.h>
#include <hip/hip_bf16.h>


typedef int   v8i __attribute__((ext_vector_type(8)));
typedef float v8f __attribute__((ext_vector_type(8)));

#define STEPS      119
#define M_ROWS     32
#define NBLOCKS    160
#define OFF_XA     262144
#define OFF_HA0    266240
#define OFF_HA1    270336
#define OFF_H1F    274432
#define OFF_PAR    290816
#define SMEM_BYTES 296960

__device__ __forceinline__ unsigned char fp8_e4m3(float x) {
  if (x != x) return (unsigned char)0x7F;
  unsigned u = __float_as_uint(x);
  unsigned s = (u >> 24) & 0x80u;
  float a = fabsf(x);
  if (a >= 448.0f) return (unsigned char)(s | 0x7Eu);
  if (a < 0.015625f) {
    int m = (int)rintf(a * 512.0f);
    if (m >= 8) return (unsigned char)(s | 0x08u);
    return (unsigned char)(s | (unsigned)m);
  }
  unsigned au = u & 0x7FFFFFFFu;
  unsigned lsb = (au >> 20) & 1u;
  au += 0x7FFFFu + lsb;
  int e = (int)(au >> 23) - 127 + 7;
  unsigned m3 = (au >> 20) & 7u;
  if (e > 15 || (e == 15 && m3 == 7)) return (unsigned char)(s | 0x7Eu);
  return (unsigned char)(s | ((unsigned)e << 3) | m3);
}

__device__ __forceinline__ float fsig(float x) { return 1.0f / (1.0f + __expf(-x)); }

__global__ void deepar_pack_weights(const float* __restrict__ Wih0,
                                    const float* __restrict__ Whh0,
                                    const float* __restrict__ Wih1,
                                    const float* __restrict__ Whh1,
                                    unsigned char* __restrict__ ws) {
  int flat  = blockIdx.x * 256 + threadIdx.x;
  int layer = flat >> 17;
  int rem   = flat & 131071;
  int n     = rem >> 8;
  int k     = rem & 255;
  float v;
  if (layer == 0)
    v = (k < 128) ? ((k < 68) ? Wih0[n * 68 + k] : 0.0f) : Whh0[n * 128 + (k - 128)];
  else
    v = (k < 128) ? Wih1[n * 128 + k] : Whh1[n * 128 + (k - 128)];
  int kk   = k & 63;
  int dest = layer * 131072
           + (((n >> 4) << 2) + (k >> 6)) * 1024
           + 32 * ((n & 15) + (((kk >> 3) & 1) << 4))
           + (((kk >> 4) << 3) + (kk & 7));
  ws[dest] = fp8_e4m3(v);
}

__device__ __forceinline__ void a_store(unsigned char* tiles, int r, int k, float v) {
  int m = r >> 4, rr = r & 15, kc = k >> 6, kk = k & 63;
  unsigned char* tb = tiles + (m * 2 + kc) * 1024;
  tb[32 * (rr + 16 * ((kk >> 3) & 1)) + 8 * (kk >> 4) + (kk & 7)] = fp8_e4m3(v);
}

__device__ __forceinline__ void lstm_layer(const unsigned char* __restrict__ wtsL,
                                           const unsigned char* __restrict__ A01,
                                           const unsigned char* __restrict__ A23,
                                           unsigned char* __restrict__ Hout,
                                           float* __restrict__ h1fbuf,
                                           const float* __restrict__ bsumL,
                                           float* __restrict__ c,
                                           int w, int lane) {
  const int col = lane & 15;
  v8f acc[2][4];
#pragma unroll
  for (int m = 0; m < 2; ++m)
#pragma unroll
    for (int g = 0; g < 4; ++g) {
      float bb = bsumL[g * 128 + w * 16 + col];
#pragma unroll
      for (int v = 0; v < 8; ++v) acc[m][g][v] = bb;
    }
#pragma unroll
  for (int kc = 0; kc < 4; ++kc) {
    const unsigned char* Ab = (kc < 2) ? (A01 + kc * 1024) : (A23 + (kc - 2) * 1024);
    v8i a0 = *(const v8i*)(Ab + 32 * lane);
    v8i a1 = *(const v8i*)(Ab + 2048 + 32 * lane);
#pragma unroll
    for (int g = 0; g < 4; ++g) {
      const v8i bt = *(const v8i*)(wtsL + ((g * 8 + w) * 4 + kc) * 1024 + 32 * lane);
      acc[0][g] = __builtin_amdgcn_wmma_f32_16x16x64_fp8_fp8(a0, bt, (short)0, acc[0][g], false, false);
      acc[1][g] = __builtin_amdgcn_wmma_f32_16x16x64_fp8_fp8(a1, bt, (short)0, acc[1][g], false, false);
    }
  }
  __syncthreads();

  const int j    = w * 16 + col;
  const int kc_h = j >> 6, kk = j & 63;
  const int hi   = (kk >> 3) & 1;
  const int boff = 8 * (kk >> 4) + (kk & 7);
#pragma unroll
  for (int m = 0; m < 2; ++m) {
    unsigned char* tb = Hout + (m * 2 + kc_h) * 1024;
#pragma unroll
    for (int v = 0; v < 8; ++v) {
      float i_ = fsig(acc[m][0][v]);
      float f_ = fsig(acc[m][1][v]);
      float g_ = tanhf(acc[m][2][v]);
      float o_ = fsig(acc[m][3][v]);
      float cc = f_ * c[m * 8 + v] + i_ * g_;
      c[m * 8 + v] = cc;
      float h = o_ * tanhf(cc);
      int rr = 8 * (lane >> 4) + v;
      tb[32 * (rr + 16 * hi) + boff] = fp8_e4m3(h);
      if (h1fbuf) h1fbuf[(m * 16 + rr) * 128 + j] = fmaxf(h, 0.0f);
    }
  }
}

__global__ void __launch_bounds__(256)
deepar_main(const float* __restrict__ hist, const float* __restrict__ fut,
            const float* __restrict__ We,  const float* __restrict__ be,
            const float* __restrict__ bih0, const float* __restrict__ bhh0,
            const float* __restrict__ bih1, const float* __restrict__ bhh1,
            const float* __restrict__ Wh,  const float* __restrict__ bh,
            const unsigned char* __restrict__ ws, float* __restrict__ out) {
  extern __shared__ unsigned char smem[];
  unsigned char* wts = smem;
  unsigned char* xA  = smem + OFF_XA;
  unsigned char* hA0 = smem + OFF_HA0;
  unsigned char* hA1 = smem + OFF_HA1;
  float* h1f  = (float*)(smem + OFF_H1F);
  float* par  = (float*)(smem + OFF_PAR);
  float* We_s = par;        float* be_s = par + 64;
  float* Wh_s = par + 128;  float* bh_s = par + 384;
  float* bsum0 = par + 386; float* bsum1 = par + 898;

  const int tid  = threadIdx.x;
  const int w    = tid >> 5;
  const int lane = tid & 31;
  const int R0   = blockIdx.x * M_ROWS;

#if defined(__gfx1250__) && __has_builtin(__builtin_amdgcn_global_load_async_to_lds_b128)
  {
    typedef int v4g __attribute__((vector_size(16)));
    typedef __attribute__((address_space(1))) v4g* gp1_t;
    typedef __attribute__((address_space(3))) v4g* lp3_t;
    for (int i = tid; i < 16384; i += 256) {
      __builtin_amdgcn_global_load_async_to_lds_b128(
          (gp1_t)(ws + 16 * i), (lp3_t)(wts + 16 * i), 0, 0);
    }
#if __has_builtin(__builtin_amdgcn_s_wait_asynccnt)
    __builtin_amdgcn_s_wait_asynccnt(0);
#else
    asm volatile("s_wait_asynccnt 0" ::: "memory");
#endif
  }
#else
  for (int i = tid; i < 16384; i += 256) ((uint4*)wts)[i] = ((const uint4*)ws)[i];
#endif

  for (int i = tid; i < 64; i += 256) { We_s[i] = We[i]; be_s[i] = be[i]; }
  for (int i = tid; i < 256; i += 256) Wh_s[i] = Wh[i];
  if (tid < 2) bh_s[tid] = bh[tid];
  for (int i = tid; i < 512; i += 256) {
    bsum0[i] = bih0[i] + bhh0[i];
    bsum1[i] = bih1[i] + bhh1[i];
  }
  for (int i = tid; i < 3072; i += 256) ((unsigned int*)(smem + OFF_XA))[i] = 0u;

  float c0[16], c1[16];
#pragma unroll
  for (int i = 0; i < 16; ++i) { c0[i] = 0.0f; c1[i] = 0.0f; }

  const int rX = tid >> 3, seg = tid & 7;
  const int grX = R0 + rX, bX = grX / 320, nX = grX % 320;
  const int rH = tid >> 1, oH = tid & 1;
  const int grH = R0 + rH, bH = grH / 320, nH = grH % 320;

  __syncthreads();

  for (int t = 0; t < STEPS; ++t) {
    {
      float ts = (t < 96) ? hist[bX * 153600 + t * 1600 + nX * 5]
                          : fut[bX * 38400 + (t - 96) * 1600 + nX * 5];
      if (seg < 4) {
#pragma unroll
        for (int q = 0; q < 16; ++q) {
          int k = seg * 16 + q;
          a_store(xA, rX, k, ts * We_s[k] + be_s[k]);
        }
      } else if (seg == 4) {
        int tc = t + 1;
#pragma unroll
        for (int ch = 1; ch <= 4; ++ch) {
          float cv = (tc < 96) ? hist[bX * 153600 + tc * 1600 + nX * 5 + ch]
                               : fut[bX * 38400 + (tc - 96) * 1600 + nX * 5 + ch];
          a_store(xA, rX, 63 + ch, cv);
        }
      }
    }
    __syncthreads();

    lstm_layer(wts,          xA,  hA0, hA0, nullptr, bsum0, c0, w, lane);
    __syncthreads();

    const bool emit = (t >= 95);
    lstm_layer(wts + 131072, hA0, hA1, hA1, emit ? h1f : nullptr, bsum1, c1, w, lane);
    __syncthreads();

    if (emit && tid < 64) {
      float a = bh_s[oH];
      const float* hr = h1f + rH * 128;
      const float* wr = Wh_s + oH * 128;
#pragma unroll 8
      for (int k = 0; k < 128; ++k) a += hr[k] * wr[k];
      float val = oH ? ((a > 20.0f) ? a : log1pf(__expf(a))) : a;
      out[((bH * 24 + (t - 95)) * 320 + nH) * 2 + oH] = val;
    }
    __syncthreads();
  }
}

extern "C" void kernel_launch(void* const* d_in, const int* in_sizes, int n_in,
                              void* d_out, int out_size, void* d_ws, size_t ws_size,
                              hipStream_t stream) {
  const float* hist = (const float*)d_in[0];
  const float* fut  = (const float*)d_in[1];
  const float* We   = (const float*)d_in[2];
  const float* be   = (const float*)d_in[3];
  const float* Wih0 = (const float*)d_in[4];
  const float* Whh0 = (const float*)d_in[5];
  const float* bih0 = (const float*)d_in[6];
  const float* bhh0 = (const float*)d_in[7];
  const float* Wih1 = (const float*)d_in[8];
  const float* Whh1 = (const float*)d_in[9];
  const float* bih1 = (const float*)d_in[10];
  const float* bhh1 = (const float*)d_in[11];
  const float* Wh   = (const float*)d_in[12];
  const float* bh   = (const float*)d_in[13];
  (void)in_sizes; (void)n_in; (void)out_size; (void)ws_size;

  (void)hipFuncSetAttribute((const void*)deepar_main,
                            hipFuncAttributeMaxDynamicSharedMemorySize,
                            (int)SMEM_BYTES);

  deepar_pack_weights<<<1024, 256, 0, stream>>>(Wih0, Whh0, Wih1, Whh1,
                                                (unsigned char*)d_ws);
  deepar_main<<<NBLOCKS, 256, SMEM_BYTES, stream>>>(
      hist, fut, We, be, bih0, bhh0, bih1, bhh1, Wh, bh,
      (const unsigned char*)d_ws, (float*)d_out);
}
